// DPBahdanauAttention_79362405695604
// MI455X (gfx1250) — matrix operand order repaired and checked by reading, not compiled, not run on hardware
//
#include <hip/hip_runtime.h>
#include <hip/hip_bf16.h>
#include <math.h>

// ---------------------------------------------------------------------------
// Bahdanau attention, MI455X (gfx1250, wave32, WMMA).
//   S=1024, B=32, QD=VD=E=1024
//   Dominant work: k = value @ Wv^T  (32768 x 1024 x 1024, 68.7 GFLOP)
//   Fused: score[s,b] = sum_e tanh(pq[b,e] + k[s,b,e]) * scale_v[e]
//   -> never materialize k (saves 137 MB of HBM round trip).
// ---------------------------------------------------------------------------

typedef __attribute__((ext_vector_type(16))) __bf16        v16bf;
typedef __attribute__((ext_vector_type(8)))  float         v8f;
typedef __attribute__((ext_vector_type(4)))  float         f4v;
typedef __attribute__((ext_vector_type(4)))  unsigned int  u4v;
typedef __attribute__((ext_vector_type(4)))  unsigned short us4v;

static constexpr int S  = 1024;
static constexpr int B  = 32;
static constexpr int QD = 1024;
static constexpr int VD = 1024;
static constexpr int E  = 1024;
static constexpr int M  = S * B;       // 32768 flattened rows of `value`
static constexpr int APAD = 8;         // LDS row pad (ushorts) to break bank conflicts

__device__ __forceinline__ unsigned short f2bf(float f) {
    unsigned u = __float_as_uint(f);
    unsigned r = u + 0x7FFFu + ((u >> 16) & 1u);   // round-to-nearest-even
    return (unsigned short)(r >> 16);
}

// ---- scale_v = (g / ||v||) * v  (one block, 1024 threads) ------------------
__global__ __launch_bounds__(1024)
void scale_v_kernel(const float* __restrict__ v, const float* __restrict__ g,
                    float* __restrict__ scale_v) {
    __shared__ float red[1024];
    const int t = threadIdx.x;
    const float x = v[t];
    red[t] = x * x;
    __syncthreads();
    for (int off = 512; off > 0; off >>= 1) {
        if (t < off) red[t] += red[t + off];
        __syncthreads();
    }
    scale_v[t] = g[0] * rsqrtf(red[0]) * x;
}

// ---- pq = query @ Wq^T + bq  (tiny GEMM, VALU is fine) ---------------------
__global__ __launch_bounds__(256)
void pq_kernel(const float* __restrict__ query, const float* __restrict__ Wq,
               const float* __restrict__ bq, float* __restrict__ pq) {
    const int e = blockIdx.x * 256 + threadIdx.x;
    const int b = blockIdx.y;
    const float* q = query + (size_t)b * QD;
    const float* w = Wq    + (size_t)e * QD;
    float acc = 0.f;
    for (int d = 0; d < QD; d += 4) {
        f4v qv = *(const f4v*)(q + d);
        f4v wv = *(const f4v*)(w + d);
        acc += qv[0]*wv[0] + qv[1]*wv[1] + qv[2]*wv[2] + qv[3]*wv[3];
    }
    pq[(size_t)b * E + e] = acc + bq[e];
}

// ---- Wv fp32 -> bf16 (kept row-major [E][VD]; 2 MB, lives in L2) ----------
__global__ __launch_bounds__(256)
void cvt_wv_kernel(const float* __restrict__ Wv, unsigned short* __restrict__ wv_bf) {
    const size_t i = ((size_t)blockIdx.x * 256 + threadIdx.x) * 4;
    f4v f = *(const f4v*)(Wv + i);
    us4v o = { f2bf(f[0]), f2bf(f[1]), f2bf(f[2]), f2bf(f[3]) };
    *(us4v*)(wv_bf + i) = o;
}

// ---- main fused GEMM + tanh + scale_v reduction ----------------------------
// One WG per 16 flattened rows of `value`. 8 waves; each wave owns 8 N-tiles
// (columns e), K loop over VD in steps of 32 with v_wmma_f32_16x16x32_bf16.
__global__ __launch_bounds__(256)
void scores_kernel(const float* __restrict__ value,
                   const unsigned short* __restrict__ wv_bf,
                   const float* __restrict__ pq,
                   const float* __restrict__ scale_v,
                   float* __restrict__ scores) {
    __shared__ unsigned short a_bf[16 * (VD + APAD)];   // ~33 KB
    __shared__ float lds_score[16];

    const int tid      = threadIdx.x;
    const int lane     = tid & 31;
    const int wave     = tid >> 5;          // 0..7
    const int h        = lane >> 4;         // lane half
    const int ln       = lane & 15;
    const int row_base = blockIdx.x * 16;

    if (tid < 16) lds_score[tid] = 0.f;

    // Stage 16 contiguous rows of value (64 KB fp32) -> bf16 in LDS.
    const float* src = value + (size_t)row_base * VD;
    for (int i = tid * 4; i < 16 * VD; i += 256 * 4) {
        f4v f = *(const f4v*)(src + i);
        us4v o = { f2bf(f[0]), f2bf(f[1]), f2bf(f[2]), f2bf(f[3]) };
        const int r = i >> 10, c = i & (VD - 1);
        *(us4v*)(a_bf + r * (VD + APAD) + c) = o;
    }
    __syncthreads();

    union F16 { v16bf v; u4v u[2]; };

    v8f acc[8];
#pragma unroll
    for (int n = 0; n < 8; ++n) acc[n] = (v8f)(0.0f);

    const unsigned short* a_row = a_bf + ln * (VD + APAD);

    for (int k0 = 0; k0 < VD; k0 += 32) {
        // A fragment: 16-bit A 16x32 layout. Lane (h,ln): M=ln,
        // K = [k0+8h .. k0+8h+7] (VGPR0-3) and [k0+16+8h ..] (VGPR4-7).
        F16 a;
        a.u[0] = *(const u4v*)(a_row + k0 + h * 8);
        a.u[1] = *(const u4v*)(a_row + k0 + 16 + h * 8);
#pragma unroll
        for (int j = 0; j < 8; ++j) {
            const int ntile = wave * 8 + j;
            // B fragment: lane half h holds K=k0+8h..k0+8h+7 and k0+16+8h..k0+16+8h+7,
            // column N = ln.  B[k][n] = Wv[n][k] -> contiguous run in wv_bf row.
            const unsigned short* bp =
                wv_bf + (size_t)(ntile * 16 + ln) * VD + k0 + h * 8;
            F16 bf;
            bf.u[0] = *(const u4v*)(bp);
            bf.u[1] = *(const u4v*)(bp + 16);
            acc[j] = __builtin_amdgcn_wmma_f32_16x16x32_bf16(
                false, a.v, false, bf.v, (short)0, acc[j], false, false);
        }
    }

    // Epilogue: C layout => VGPR i holds (M = i + 8h, N = ln) of the tile.
    float partial[8];
#pragma unroll
    for (int i = 0; i < 8; ++i) partial[i] = 0.f;

#pragma unroll
    for (int j = 0; j < 8; ++j) {
        const int col = (wave * 8 + j) * 16 + ln;       // global e
        const float sv = scale_v[col];
#pragma unroll
        for (int i = 0; i < 8; ++i) {
            const int m = i + h * 8;                    // row within tile
            const int b = (row_base + m) & (B - 1);     // flattened row % B
            partial[i] += tanhf(pq[(size_t)b * E + col] + acc[j][i]) * sv;
        }
    }

    // Reduce over the 16 columns (lanes) of each half, then across waves in LDS.
#pragma unroll
    for (int i = 0; i < 8; ++i) {
        float p = partial[i];
        p += __shfl_xor(p, 1, 32);
        p += __shfl_xor(p, 2, 32);
        p += __shfl_xor(p, 4, 32);
        p += __shfl_xor(p, 8, 32);
        if (ln == 0) atomicAdd(&lds_score[i + h * 8], p);   // ds_add_f32
    }
    __syncthreads();
    if (tid < 16) scores[row_base + tid] = lds_score[tid];
}

// ---- softmax over S (per batch column), apply key_padding_mask -------------
__global__ __launch_bounds__(1024)
void softmax_kernel(const float* __restrict__ scores,
                    const unsigned char* __restrict__ mask,
                    float* __restrict__ probs,
                    float* __restrict__ out_attn0,
                    float* __restrict__ out_attn1) {
    __shared__ float red[1024];
    const int s = threadIdx.x;
    const int b = blockIdx.x;
    const int idx = s * B + b;
    float x = scores[idx];
    if (mask[idx]) x = -INFINITY;

    red[s] = x;
    __syncthreads();
    for (int off = 512; off > 0; off >>= 1) {
        if (s < off) red[s] = fmaxf(red[s], red[s + off]);
        __syncthreads();
    }
    const float mx = red[0];
    __syncthreads();

    const float e = __expf(x - mx);
    red[s] = e;
    __syncthreads();
    for (int off = 512; off > 0; off >>= 1) {
        if (s < off) red[s] += red[s + off];
        __syncthreads();
    }
    const float p = e / red[0];
    probs[idx]     = p;
    out_attn0[idx] = p;
    out_attn1[idx] = p;
}

// ---- context[b,d] = sum_s p[s,b] * value[s,b,d]  (L2-resident 2nd pass) ----
__global__ __launch_bounds__(256)
void context_kernel(const float* __restrict__ value,
                    const float* __restrict__ probs,
                    float* __restrict__ context) {
    const int d = blockIdx.x * 256 + threadIdx.x;
    const int b = blockIdx.y;
    const float* vp = value + (size_t)b * VD + d;
    float acc = 0.f;
    for (int s = 0; s < S; ++s) {
        __builtin_prefetch(vp + (size_t)(s + 8) * B * VD, 0, 0);  // global_prefetch_b8
        acc += probs[s * B + b] * vp[(size_t)s * B * VD];
    }
    context[(size_t)b * VD + d] = acc;
}

// ---------------------------------------------------------------------------
extern "C" void kernel_launch(void* const* d_in, const int* in_sizes, int n_in,
                              void* d_out, int out_size, void* d_ws, size_t ws_size,
                              hipStream_t stream) {
    const float*         query = (const float*)d_in[0];
    const float*         value = (const float*)d_in[1];
    const unsigned char* mask  = (const unsigned char*)d_in[2];  // jnp.bool_ = 1 byte
    const float*         Wq    = (const float*)d_in[3];
    const float*         bq    = (const float*)d_in[4];
    const float*         Wv    = (const float*)d_in[5];
    const float*         v     = (const float*)d_in[6];
    const float*         g     = (const float*)d_in[7];

    float* out          = (float*)d_out;
    float* out_context  = out;                   // [B, VD]
    float* out_attn0    = out + B * VD;          // [S, B]
    float* out_attn1    = out + B * VD + S * B;  // [S, B] (duplicate return)

    // Workspace layout (~2.4 MB total)
    float* ws_pq     = (float*)d_ws;             // B*E
    float* ws_scale  = ws_pq + B * E;            // E
    float* ws_scores = ws_scale + E;             // M
    float* ws_probs  = ws_scores + M;            // M
    unsigned short* ws_wv = (unsigned short*)(ws_probs + M);  // E*VD bf16

    scale_v_kernel<<<1, E, 0, stream>>>(v, g, ws_scale);
    pq_kernel<<<dim3(E / 256, B), 256, 0, stream>>>(query, Wq, bq, ws_pq);
    cvt_wv_kernel<<<((size_t)E * VD / 4) / 256, 256, 0, stream>>>(Wv, ws_wv);
    scores_kernel<<<M / 16, 256, 0, stream>>>(value, ws_wv, ws_pq, ws_scale, ws_scores);
    softmax_kernel<<<B, S, 0, stream>>>(ws_scores, mask, ws_probs, out_attn0, out_attn1);
    context_kernel<<<dim3(VD / 256, B), 256, 0, stream>>>(value, ws_probs, out_context);
}
